// MultiHeadAttention_16071767622128
// MI455X (gfx1250) — matrix operand order repaired and checked by reading, not compiled, not run on hardware
//
#include <hip/hip_runtime.h>

// ---------- CDNA5 types ----------
typedef __attribute__((ext_vector_type(16))) __bf16        v16bf;
typedef __attribute__((ext_vector_type(8)))  float         v8f;
typedef __attribute__((ext_vector_type(4)))  unsigned int  u32x4;
typedef __attribute__((ext_vector_type(4)))  float         f32x4;

union Frag {
    v16bf        bf;
    unsigned int u[8];
};

__device__ __forceinline__ unsigned short f2bf(float f) {
    unsigned int u = __builtin_bit_cast(unsigned int, f);
    u += 0x7FFFu + ((u >> 16) & 1u);          // round-to-nearest-even
    return (unsigned short)(u >> 16);
}

// Async global->LDS copy, 16 bytes per lane (tracked by ASYNCcnt).
__device__ __forceinline__ void async_ld_b128(const void* gptr, void* lptr) {
    unsigned lds          = (unsigned)(uintptr_t)lptr;   // low 32 bits = LDS offset
    unsigned long long ga = (unsigned long long)(uintptr_t)gptr;
    asm volatile("global_load_async_to_lds_b128 %0, %1, off"
                 :: "v"(lds), "v"(ga) : "memory");
}
__device__ __forceinline__ void wait_async0() {
    asm volatile("s_wait_asynccnt 0x0" ::: "memory");
}

__device__ __forceinline__ void ld_frag_lds(Frag& fr, const unsigned short* p0,
                                            const unsigned short* p1) {
    u32x4 lo = *(const u32x4*)p0;
    u32x4 hi = *(const u32x4*)p1;
    fr.u[0] = lo[0]; fr.u[1] = lo[1]; fr.u[2] = lo[2]; fr.u[3] = lo[3];
    fr.u[4] = hi[0]; fr.u[5] = hi[1]; fr.u[6] = hi[2]; fr.u[7] = hi[3];
}

__device__ __forceinline__ v8f wmma_bf16(const Frag& a, const Frag& b, v8f c) {
    return __builtin_amdgcn_wmma_f32_16x16x32_bf16(
        false, a.bf, false, b.bf, (short)0, c, false, false);
}

// =====================================================================
// Transpose + convert: Wt[n][k] = bf16(W[k][n]).  rows=K, cols=N.
// =====================================================================
__global__ __launch_bounds__(256)
void transpose_f2bf(const float* __restrict__ W, unsigned short* __restrict__ Wt,
                    int rows, int cols)
{
    __shared__ float tile[32][33];
    const int bx = blockIdx.x * 32;            // col base (n)
    const int by = blockIdx.y * 32;            // row base (k)
    const int tx = threadIdx.x & 31;
    const int ty = threadIdx.x >> 5;           // 0..7
    #pragma unroll
    for (int j = 0; j < 32; j += 8)
        tile[ty + j][tx] = W[(size_t)(by + ty + j) * cols + bx + tx];
    __syncthreads();
    #pragma unroll
    for (int j = 0; j < 32; j += 8)
        Wt[(size_t)(bx + ty + j) * rows + by + tx] = f2bf(tile[tx][ty + j]);
}

// =====================================================================
// GEMM:  Out[M,N] = scale * (A[M,K] @ Wt^T + bias),  Wt is [N][K] bf16.
// IN_BF16: A is bf16 (async path) else f32 (convert in regs).
// OUT_MODE: 0 = f32 row-major, 1 = bf16 row-major,
//           2 = bf16 transposed per batch: out[(b*1024+col)*2048 + t]
// BM=128, BN=64, BK=64 ; 256 threads = 8 waves ; wave: 16x64 strip.
// Double-buffered LDS: async loads for tile i+1 overlap WMMAs on tile i.
// =====================================================================
template<bool IN_BF16, int OUT_MODE>
__global__ __launch_bounds__(256)
void gemm_bf16_wmma(const void* __restrict__ Ap,
                    const unsigned short* __restrict__ Wt,
                    const float* __restrict__ bias, void* __restrict__ Outp,
                    int M, int N, int K, float out_scale)
{
    __shared__ unsigned short As[2][128][64];  // [buf][m][k] bf16 (32 KB)
    __shared__ unsigned short Bs[2][64][64];   // [buf][n][k] bf16 (16 KB)

    const int tid   = threadIdx.x;
    const int lane  = tid & 31;
    const int wv    = tid >> 5;
    const int lrow  = lane & 15;
    const int khalf = lane >> 4;
    const int gm0   = blockIdx.x * 128;
    const int gn0   = blockIdx.y * 64;

    v8f acc[4] = {v8f{}, v8f{}, v8f{}, v8f{}};

    auto issue_tiles = [&](int kk, int p) {
        // B tile: 64x64 bf16 -> 2 x b128 per thread (async)
        #pragma unroll
        for (int j = 0; j < 2; ++j) {
            const int idx = tid * 2 + j;
            const int n   = idx >> 3;
            const int c8  = (idx & 7) * 8;
            async_ld_b128(Wt + (size_t)(gn0 + n) * K + kk + c8, &Bs[p][n][c8]);
        }
        // A tile: 128x64
        if (IN_BF16) {                          // async, 4 x b128 per thread
            const unsigned short* A = (const unsigned short*)Ap;
            #pragma unroll
            for (int j = 0; j < 4; ++j) {
                const int idx = tid * 4 + j;
                const int r   = idx >> 3;
                const int c8  = (idx & 7) * 8;
                async_ld_b128(A + (size_t)(gm0 + r) * K + kk + c8, &As[p][r][c8]);
            }
        } else {                                // f32 -> bf16 in registers
            const float* A = (const float*)Ap;
            #pragma unroll
            for (int j = 0; j < 8; ++j) {
                const int idx = tid * 8 + j;
                const int r   = idx >> 4;
                const int c4  = (idx & 15) * 4;
                f32x4 v = *(const f32x4*)(A + (size_t)(gm0 + r) * K + kk + c4);
                unsigned short* d = &As[p][r][c4];
                d[0] = f2bf(v[0]); d[1] = f2bf(v[1]);
                d[2] = f2bf(v[2]); d[3] = f2bf(v[3]);
            }
        }
    };

    issue_tiles(0, 0);
    wait_async0();
    __syncthreads();

    int p = 0;
    for (int kk = 0; kk < K; kk += 64) {
        if (kk + 64 < K) issue_tiles(kk + 64, p ^ 1);   // overlap with compute

        #pragma unroll
        for (int f = 0; f < 2; ++f) {
            Frag a;
            const unsigned short* pa = &As[p][wv * 16 + lrow][f * 32 + khalf * 8];
            ld_frag_lds(a, pa, pa + 16);        // A: K in {kh*8..+7, 16+kh*8..+7}
            #pragma unroll
            for (int j = 0; j < 4; ++j) {
                Frag bfr;
                const unsigned short* pb = &Bs[p][j * 16 + lrow][f * 32 + khalf * 8];
                ld_frag_lds(bfr, pb, pb + 16);  // B: K in {kh*8..+7, 16+kh*8..+7}
                acc[j] = wmma_bf16(a, bfr, acc[j]);
            }
        }
        wait_async0();                          // next tiles landed
        __syncthreads();                        // everyone done reading buf p
        p ^= 1;
    }

    // ---- epilogue ----
    #pragma unroll
    for (int j = 0; j < 4; ++j) {
        const int col = gn0 + j * 16 + lrow;
        const float bv = bias[col];
        #pragma unroll
        for (int i = 0; i < 8; ++i) {
            const int row   = gm0 + wv * 16 + i + 8 * khalf;   // b*2048 + t
            const float val = (acc[j][i] + bv) * out_scale;
            if (OUT_MODE == 0) {
                ((float*)Outp)[(size_t)row * N + col] = val;
            } else if (OUT_MODE == 1) {
                ((unsigned short*)Outp)[(size_t)row * N + col] = f2bf(val);
            } else {                                           // transposed (V)
                const size_t off = ((size_t)(row >> 11) * 1024 + col) * 2048
                                   + (row & 2047);
                ((unsigned short*)Outp)[off] = f2bf(val);
            }
        }
    }
}

// =====================================================================
// Flash attention (causal), bf16 WMMA, online softmax, double-buffered
// async K/V tiles.  grid = (T/128, H, B), block = 256 (8 waves).
// Q pre-scaled by 1/sqrt(64); V supplied transposed: Vt[b*1024+d'][t].
// =====================================================================
__global__ __launch_bounds__(256)
void flash_attn_wmma(const unsigned short* __restrict__ Qb,
                     const unsigned short* __restrict__ Kb,
                     const unsigned short* __restrict__ Vt,
                     unsigned short* __restrict__ Ctx)
{
    __shared__ unsigned short Ks[2][32][64];     // [buf][s][dk]
    __shared__ unsigned short Vs[2][64][32];     // [buf][dv][s]
    __shared__ unsigned short Ps[8][16][40];     // per-wave P staging (padded)

    const int tid   = threadIdx.x;
    const int lane  = tid & 31;
    const int wv    = tid >> 5;
    const int lrow  = lane & 15;
    const int khalf = lane >> 4;
    const int tBase = blockIdx.x * 128;
    const int h     = blockIdx.y;
    const int b     = blockIdx.z;

    const size_t ld       = 1024;
    const size_t batchOff = (size_t)b * 2048 * ld;
    const size_t headOff  = (size_t)h * 64;
    const unsigned short* vtp = Vt + ((size_t)b * 1024 + h * 64) * 2048;

    // ---- Q fragments in registers (16 rows x 64 dk = 2 A-frags) ----
    Frag qf[2];
    {
        const unsigned short* qp =
            Qb + batchOff + (size_t)(tBase + wv * 16 + lrow) * ld + headOff;
        #pragma unroll
        for (int f = 0; f < 2; ++f) {
            u32x4 lo = *(const u32x4*)(qp + f * 32 + khalf * 8);
            u32x4 hi = *(const u32x4*)(qp + f * 32 + 16 + khalf * 8);
            qf[f].u[0] = lo[0]; qf[f].u[1] = lo[1]; qf[f].u[2] = lo[2]; qf[f].u[3] = lo[3];
            qf[f].u[4] = hi[0]; qf[f].u[5] = hi[1]; qf[f].u[6] = hi[2]; qf[f].u[7] = hi[3];
        }
    }

    v8f o[4] = {v8f{}, v8f{}, v8f{}, v8f{}};
    float mrow[8], lsum[8];
    #pragma unroll
    for (int i = 0; i < 8; ++i) { mrow[i] = -3.0e38f; lsum[i] = 0.0f; }

    auto issue_kv = [&](int s0, int p) {
        const int r  = tid >> 3;                 // K: s row 0..31
        const int c8 = (tid & 7) * 8;
        async_ld_b128(Kb + batchOff + (size_t)(s0 + r) * ld + headOff + c8,
                      &Ks[p][r][c8]);
        const int d  = tid >> 2;                 // V: d row 0..63
        const int v8c = (tid & 3) * 8;
        async_ld_b128(vtp + (size_t)d * 2048 + s0 + v8c, &Vs[p][d][v8c]);
    };

    const int sEnd = tBase + 128;
    issue_kv(0, 0);
    wait_async0();
    __syncthreads();

    int p = 0;
    for (int s0 = 0; s0 < sEnd; s0 += 32) {
        if (s0 + 32 < sEnd) issue_kv(s0 + 32, p ^ 1);   // overlap with compute

        if (s0 <= tBase + wv * 16 + 15) {        // wave-uniform causal skip
            // ---- scores ----
            v8f sc[2];
            #pragma unroll
            for (int j = 0; j < 2; ++j) {
                v8f s = {};
                #pragma unroll
                for (int f = 0; f < 2; ++f) {
                    Frag kb;
                    const unsigned short* pk = &Ks[p][j * 16 + lrow][f * 32 + khalf * 8];
                    ld_frag_lds(kb, pk, pk + 16);
                    s = wmma_bf16(qf[f], kb, s);
                }
                const int scol = s0 + j * 16 + lrow;
                #pragma unroll
                for (int i = 0; i < 8; ++i) {
                    const int trow = tBase + wv * 16 + i + 8 * khalf;
                    if (scol > trow) s[i] = -1.0e10f;
                }
                sc[j] = s;
            }
            // ---- online softmax (row reduce over 16-lane half) ----
            float alpha[8];
            #pragma unroll
            for (int i = 0; i < 8; ++i) {
                float mt = fmaxf(sc[0][i], sc[1][i]);
                #pragma unroll
                for (int msk = 1; msk < 16; msk <<= 1)
                    mt = fmaxf(mt, __shfl_xor(mt, msk, 32));
                const float mNew = fmaxf(mrow[i], mt);
                alpha[i] = __expf(mrow[i] - mNew);
                mrow[i]  = mNew;
            }
            #pragma unroll
            for (int i = 0; i < 8; ++i) {
                sc[0][i] = __expf(sc[0][i] - mrow[i]);
                sc[1][i] = __expf(sc[1][i] - mrow[i]);
                float r = sc[0][i] + sc[1][i];
                #pragma unroll
                for (int msk = 1; msk < 16; msk <<= 1)
                    r += __shfl_xor(r, msk, 32);
                lsum[i] = lsum[i] * alpha[i] + r;
            }
            #pragma unroll
            for (int jt = 0; jt < 4; ++jt)
                #pragma unroll
                for (int i = 0; i < 8; ++i) o[jt][i] *= alpha[i];

            // ---- P -> LDS (reshape into A-fragment layout) ----
            #pragma unroll
            for (int j = 0; j < 2; ++j)
                #pragma unroll
                for (int i = 0; i < 8; ++i)
                    Ps[wv][i + 8 * khalf][j * 16 + lrow] = f2bf(sc[j][i]);

            Frag pf;
            {
                const unsigned short* pp = &Ps[wv][lrow][khalf * 8];
                ld_frag_lds(pf, pp, pp + 16);
            }
            // ---- PV ----
            #pragma unroll
            for (int jt = 0; jt < 4; ++jt) {
                Frag vf;
                const unsigned short* pv = &Vs[p][jt * 16 + lrow][khalf * 8];
                ld_frag_lds(vf, pv, pv + 16);
                o[jt] = wmma_bf16(pf, vf, o[jt]);
            }
        }
        wait_async0();                           // next K/V tiles landed
        __syncthreads();                         // all waves done with buf p
        p ^= 1;
    }

    // ---- finalize: O / l -> ctx (bf16, row-major [b*2048+t][h*64+d]) ----
    unsigned short* cp = Ctx + batchOff + headOff;
    #pragma unroll
    for (int i = 0; i < 8; ++i) {
        const float inv = 1.0f / lsum[i];
        const size_t r  = (size_t)(tBase + wv * 16 + i + 8 * khalf) * ld;
        #pragma unroll
        for (int jt = 0; jt < 4; ++jt)
            cp[r + jt * 16 + lrow] = f2bf(o[jt][i] * inv);
    }
}

// =====================================================================
// Host launcher
// =====================================================================
extern "C" void kernel_launch(void* const* d_in, const int* in_sizes, int n_in,
                              void* d_out, int out_size, void* d_ws, size_t ws_size,
                              hipStream_t stream) {
    (void)in_sizes; (void)n_in; (void)out_size; (void)ws_size;
    const float* query = (const float*)d_in[0];
    const float* key   = (const float*)d_in[1];
    const float* value = (const float*)d_in[2];
    const float* Wq    = (const float*)d_in[3];
    const float* bq    = (const float*)d_in[4];
    const float* Wk    = (const float*)d_in[5];
    const float* bk    = (const float*)d_in[6];
    const float* Wv    = (const float*)d_in[7];
    const float* bv    = (const float*)d_in[8];
    const float* Wo    = (const float*)d_in[9];
    const float* bo    = (const float*)d_in[10];

    const int B = 4, T = 2048, D = 1024;
    const int M = B * T;                          // 8192
    const size_t wsz = (size_t)D * D;             // 1M elems per weight
    const size_t msz = (size_t)M * D;             // 8M elems per activation

    unsigned short* Wtq = (unsigned short*)d_ws;
    unsigned short* Wtk = Wtq + wsz;
    unsigned short* Wtv = Wtk + wsz;
    unsigned short* Wto = Wtv + wsz;
    unsigned short* Qb  = Wto + wsz;
    unsigned short* Kb  = Qb  + msz;
    unsigned short* Vt  = Kb  + msz;
    unsigned short* Ctx = Vt  + msz;

    dim3 blk(256);
    dim3 gt(D / 32, D / 32);                      // (32, 32)
    transpose_f2bf<<<gt, blk, 0, stream>>>(Wq, Wtq, D, D);
    transpose_f2bf<<<gt, blk, 0, stream>>>(Wk, Wtk, D, D);
    transpose_f2bf<<<gt, blk, 0, stream>>>(Wv, Wtv, D, D);
    transpose_f2bf<<<gt, blk, 0, stream>>>(Wo, Wto, D, D);

    dim3 gg(M / 128, D / 64);                     // (64, 16)
    gemm_bf16_wmma<false, 1><<<gg, blk, 0, stream>>>(query, Wtq, bq, Qb, M, D, D, 0.125f);
    gemm_bf16_wmma<false, 1><<<gg, blk, 0, stream>>>(key,   Wtk, bk, Kb, M, D, D, 1.0f);
    gemm_bf16_wmma<false, 2><<<gg, blk, 0, stream>>>(value, Wtv, bv, Vt, M, D, D, 1.0f);

    dim3 ga(T / 128, 16, B);                      // (16, 16, 4)
    flash_attn_wmma<<<ga, blk, 0, stream>>>(Qb, Kb, Vt, Ctx);

    gemm_bf16_wmma<true, 0><<<gg, blk, 0, stream>>>(Ctx, Wto, bo, (float*)d_out, M, D, D, 1.0f);
}
